// GraphTGI_hetero_57200374448676
// MI455X (gfx1250) — matrix operand order repaired and checked by reading, not compiled, not run on hardware
//
#include <hip/hip_runtime.h>
#include <stdint.h>

#define N_TF 10000
#define N_TG 40000
#define N_MI 5000
#define E_EDGES 250000
#define D_IN 768
#define D1 1024
#define D2 512
#define D3 256

typedef _Float16 h16;
typedef h16  v16h __attribute__((ext_vector_type(16)));
typedef __fp16 fp16x2 __attribute__((ext_vector_type(2)));   // cvt_pkrtz result type
typedef float v8f  __attribute__((ext_vector_type(8)));

union FragU { v16h v; uint32_t u[8]; };
union PackU { fp16x2 h[2]; uint2 u2; };

// ---------------------------------------------------------------------------
// Generic f16-WMMA GEMM: out = act(A[MxK] @ W[KxN] + bias + addend), with
// optional '+=' accumulation into out. N % 64 == 0, K % 32 == 0, M arbitrary.
// Block tile 128x64, 8 waves (wave32), each wave a 32x32 tile = 2x2 WMMA.
// Software pipeline: b128 global loads of tile k+1 overlap WMMAs of tile k.
// ---------------------------------------------------------------------------
__global__ __launch_bounds__(256)
void gemm_wmma_f16(const float* __restrict__ A, const float* __restrict__ W,
                   const float* __restrict__ bias, const float* __restrict__ addend,
                   float* __restrict__ out, int M, int N, int K,
                   int act, int accum)
{
    __shared__ h16 lds_a[128][36];   // 72B row stride: 8B-aligned packed stores,
    __shared__ h16 lds_bt[64][36];   // conflict-free fragment reads (gcd(18,64)=2)

    const int tid  = threadIdx.x;
    const int lane = tid & 31;
    const int wave = tid >> 5;
    const int wr   = (wave & 3) * 32;   // wave row offset in block tile
    const int wc   = (wave >> 2) * 32;  // wave col offset in block tile
    const int m0   = blockIdx.y * 128;
    const int n0   = blockIdx.x * 64;
    const int h    = lane >> 4;         // lane half (ISA 16-bit frag layout)
    const int ln   = lane & 15;

    // Staging coordinates: A = 4 float4/thread (128x32), B = 2 float4/thread (32x64)
    int a_m[4], a_k[4], b_k[2], b_n[2];
    #pragma unroll
    for (int i = 0; i < 4; ++i) { int f = tid + i * 256; a_m[i] = f >> 3; a_k[i] = (f & 7) * 4; }
    #pragma unroll
    for (int i = 0; i < 2; ++i) { int f = tid + i * 256; b_k[i] = f >> 4; b_n[i] = (f & 15) * 4; }

    float4 ra[4], rb[2];

    auto gload = [&](int k0) {
        #pragma unroll
        for (int i = 0; i < 4; ++i) {
            int gm  = m0 + a_m[i];
            int gmc = (gm < M) ? gm : (M - 1);     // clamp: unconditional b128 load
            ra[i] = *(const float4*)(A + (size_t)gmc * K + (k0 + a_k[i]));
            if (gm >= M) ra[i] = make_float4(0.f, 0.f, 0.f, 0.f);
        }
        #pragma unroll
        for (int i = 0; i < 2; ++i)
            rb[i] = *(const float4*)(W + (size_t)(k0 + b_k[i]) * N + (n0 + b_n[i]));
    };

    auto lstore = [&]() {
        #pragma unroll
        for (int i = 0; i < 4; ++i) {
            PackU pk;
            pk.h[0] = __builtin_amdgcn_cvt_pkrtz(ra[i].x, ra[i].y);
            pk.h[1] = __builtin_amdgcn_cvt_pkrtz(ra[i].z, ra[i].w);
            *(uint2*)&lds_a[a_m[i]][a_k[i]] = pk.u2;   // 8B-aligned -> ds_store_b64
        }
        #pragma unroll
        for (int i = 0; i < 2; ++i) {                  // transposed: [n][k]
            lds_bt[b_n[i] + 0][b_k[i]] = (h16)rb[i].x;
            lds_bt[b_n[i] + 1][b_k[i]] = (h16)rb[i].y;
            lds_bt[b_n[i] + 2][b_k[i]] = (h16)rb[i].z;
            lds_bt[b_n[i] + 3][b_k[i]] = (h16)rb[i].w;
        }
    };

    v8f acc[2][2];
    #pragma unroll
    for (int i = 0; i < 2; ++i)
        #pragma unroll
        for (int j = 0; j < 2; ++j)
            #pragma unroll
            for (int v = 0; v < 8; ++v)
                acc[i][j][v] = 0.0f;

    gload(0);
    for (int k0 = 0; k0 < K; k0 += 32) {
        lstore();
        __syncthreads();
        if (k0 + 32 < K) gload(k0 + 32);   // next tile in flight during WMMAs

        // A fragment (16x32 f16): lane = row; elems 0-7 -> K = 8h+0..7,
        // elems 8-15 -> K = 16+8h+0..7  (ISA 7.12.2, 16-bit A layout)
        v16h afrag[2], bfrag[2];
        #pragma unroll
        for (int mi = 0; mi < 2; ++mi) {
            FragU fa;
            const uint32_t* p0 = (const uint32_t*)&lds_a[wr + mi * 16 + ln][h * 8];
            const uint32_t* p1 = (const uint32_t*)&lds_a[wr + mi * 16 + ln][16 + h * 8];
            #pragma unroll
            for (int j = 0; j < 4; ++j) { fa.u[j] = p0[j]; fa.u[4 + j] = p1[j]; }
            afrag[mi] = fa.v;
        }
        // B fragment (32x16 f16): lane = col; elems 0-7 -> K = 8h+0..7, elems 8-15 -> K = 16+8h+0..7
        #pragma unroll
        for (int ni = 0; ni < 2; ++ni) {
            FragU fb;
            const uint32_t* p0 = (const uint32_t*)&lds_bt[wc + ni * 16 + ln][h * 8];
            const uint32_t* p1 = (const uint32_t*)&lds_bt[wc + ni * 16 + ln][16 + h * 8];
            #pragma unroll
            for (int j = 0; j < 4; ++j) { fb.u[j] = p0[j]; fb.u[4 + j] = p1[j]; }
            bfrag[ni] = fb.v;
        }

        #pragma unroll
        for (int mi = 0; mi < 2; ++mi)
            #pragma unroll
            for (int ni = 0; ni < 2; ++ni)
                acc[mi][ni] = __builtin_amdgcn_wmma_f32_16x16x32_f16(
                    false, afrag[mi], false, bfrag[ni],
                    (short)0, acc[mi][ni], false, false);
        __syncthreads();
    }

    // Store: C/D layout -> VGPR v, half h holds row (8h+v), col = lane&15
    #pragma unroll
    for (int mi = 0; mi < 2; ++mi) {
        #pragma unroll
        for (int ni = 0; ni < 2; ++ni) {
            int gc = n0 + wc + ni * 16 + ln;
            #pragma unroll
            for (int v = 0; v < 8; ++v) {
                int gr = m0 + wr + mi * 16 + h * 8 + v;
                if (gr < M) {
                    float val = acc[mi][ni][v];
                    if (bias)   val += bias[gc];
                    if (addend) val += addend[(size_t)gr * N + gc];
                    if (act == 1)      val = fmaxf(val, 0.0f);
                    else if (act == 2) val = (val > 0.0f) ? val : 0.2f * val;
                    size_t o = (size_t)gr * N + gc;
                    if (accum) out[o] += val;
                    else       out[o] = val;
                }
            }
        }
    }
}

// ---------------------------------------------------------------------------
// Segment-max over edges: one block per edge, float4 per thread.
// pooled >= 0 (ReLU output), so integer atomicMax on the bit pattern is exact
// and a zero-initialized agg reproduces segment_max's -inf -> 0 fixup.
// ---------------------------------------------------------------------------
__global__ void seg_max_kernel(const float* __restrict__ pooled, const int* __restrict__ src,
                               const int* __restrict__ dst, float* __restrict__ agg, int D)
{
    int e = blockIdx.x;
    int c = threadIdx.x << 2;
    int s = src[e], d = dst[e];
    const float4 v = *reinterpret_cast<const float4*>(pooled + (size_t)s * D + c);
    int* o = reinterpret_cast<int*>(agg + (size_t)d * D + c);
    atomicMax(o + 0, __float_as_int(v.x));
    atomicMax(o + 1, __float_as_int(v.y));
    atomicMax(o + 2, __float_as_int(v.z));
    atomicMax(o + 3, __float_as_int(v.w));
}

__global__ void fill0(float* __restrict__ p, long long n4)
{
    long long i = (long long)blockIdx.x * blockDim.x + threadIdx.x;
    if (i < n4) reinterpret_cast<float4*>(p)[i] = make_float4(0.f, 0.f, 0.f, 0.f);
}

__global__ void leaky_ip(float* __restrict__ p, long long n)
{
    long long i = (long long)blockIdx.x * blockDim.x + threadIdx.x;
    if (i < n) { float v = p[i]; p[i] = (v > 0.0f) ? v : 0.2f * v; }
}

// ---------------------------------------------------------------------------
static inline void launch_gemm(const float* A, const float* W, const float* bias,
                               const float* addend, float* out, int M, int N, int K,
                               int act, int accum, hipStream_t s)
{
    dim3 grid(N / 64, (M + 127) / 128);
    gemm_wmma_f16<<<grid, 256, 0, s>>>(A, W, bias, addend, out, M, N, K, act, accum);
}

static inline void launch_fill0(float* p, long long n, hipStream_t s)
{
    long long n4 = n >> 2;  // all buffer sizes are multiples of 4
    fill0<<<(unsigned)((n4 + 255) / 256), 256, 0, s>>>(p, n4);
}

extern "C" void kernel_launch(void* const* d_in, const int* in_sizes, int n_in,
                              void* d_out, int out_size, void* d_ws, size_t ws_size,
                              hipStream_t stream)
{
    (void)in_sizes; (void)n_in; (void)out_size; (void)ws_size;

    const float* x_tf     = (const float*)d_in[0];
    const float* x_tg     = (const float*)d_in[1];
    const float* x_mi     = (const float*)d_in[2];
    const float* W_emb_tf = (const float*)d_in[3];
    const float* b_emb_tf = (const float*)d_in[4];
    const float* W_emb_tg = (const float*)d_in[5];
    const float* b_emb_tg = (const float*)d_in[6];
    const float* W_emb_mi = (const float*)d_in[7];
    const float* b_emb_mi = (const float*)d_in[8];
    const float* Wp1 = (const float*)d_in[9];
    const float* bp1 = (const float*)d_in[10];
    const float* Wn1 = (const float*)d_in[11];
    const float* Ws1 = (const float*)d_in[12];
    const float* bs1 = (const float*)d_in[13];
    const float* Wp2 = (const float*)d_in[14];
    const float* bp2 = (const float*)d_in[15];
    const float* Wn2 = (const float*)d_in[16];
    const float* Ws2 = (const float*)d_in[17];
    const float* bs2 = (const float*)d_in[18];
    const int* srcA[6]; const int* dstA[6];
    for (int i = 0; i < 6; ++i) {
        srcA[i] = (const int*)d_in[19 + 2 * i];
        dstA[i] = (const int*)d_in[20 + 2 * i];
    }

    // Workspace layout (floats): ~166.4M floats
    float* w = (float*)d_ws;
    size_t off = 0;
    float* h_tf = w + off; off += (size_t)N_TF * D1;
    float* h_tg = w + off; off += (size_t)N_TG * D1;
    float* h_mi = w + off; off += (size_t)N_MI * D1;
    float* g_tf = w + off; off += (size_t)N_TF * D2;
    float* g_tg = w + off; off += (size_t)N_TG * D2;
    float* g_mi = w + off; off += (size_t)N_MI * D2;
    float* pool = w + off; off += (size_t)N_TG * D1;   // pooled; reused as self-temp t
    float* agg  = w + off; off += (size_t)N_TG * D1;

    // ---- Embeddings: h = x @ W_emb + b ----
    launch_gemm(x_tf, W_emb_tf, b_emb_tf, nullptr, h_tf, N_TF, D1, D_IN, 0, 0, stream);
    launch_gemm(x_tg, W_emb_tg, b_emb_tg, nullptr, h_tg, N_TG, D1, D_IN, 0, 0, stream);
    launch_gemm(x_mi, W_emb_mi, b_emb_mi, nullptr, h_mi, N_MI, D1, D_IN, 0, 0, stream);

    // ---- Hetero SAGE layer ----
    auto run_layer = [&](const float* Wp, const float* bp, const float* Wn,
                         const float* Ws, const float* bs, int Din, int Dout,
                         float* htf, float* htg, float* hmi,
                         float* otf, float* otg, float* omi)
    {
        const float* hs[6]  = {htf, htg, htf, hmi, hmi, htf};
        const float* hd[6]  = {htg, htf, hmi, htf, htg, htf};
        float*       oa[6]  = {otg, otf, omi, otf, otg, otf};
        const int    ns[6]  = {N_TF, N_TG, N_TF, N_MI, N_MI, N_TF};
        const int    nd[6]  = {N_TG, N_TF, N_MI, N_TF, N_TG, N_TF};

        for (int i = 0; i < 6; ++i) {
            const float* Wpi = Wp + (size_t)i * Din * Din;
            const float* bpi = bp + (size_t)i * Din;
            const float* Wni = Wn + (size_t)i * Din * Dout;
            const float* Wsi = Ws + (size_t)i * Din * Dout;
            const float* bsi = bs + (size_t)i * Dout;

            // pooled = relu(h_src @ Wp + bp)
            launch_gemm(hs[i], Wpi, bpi, nullptr, pool, ns[i], Din, Din, 1, 0, stream);
            // agg = segment_max(pooled[src], dst) with zero floor
            launch_fill0(agg, (long long)nd[i] * Din, stream);
            seg_max_kernel<<<E_EDGES, Din / 4, 0, stream>>>(pool, srcA[i], dstA[i], agg, Din);
            // t = h_dst @ Ws + bs      (t aliases pool; pooled is dead now)
            float* t = pool;
            launch_gemm(hd[i], Wsi, bsi, nullptr, t, nd[i], Dout, Din, 0, 0, stream);
            // out_acc += leaky_relu(t + agg @ Wn)
            launch_gemm(agg, Wni, nullptr, t, oa[i], nd[i], Dout, Din, 2, 1, stream);
        }
    };

    // Layer 1 -> g buffers (zeroed accumulation targets)
    launch_fill0(g_tf, (long long)N_TF * D2, stream);
    launch_fill0(g_tg, (long long)N_TG * D2, stream);
    launch_fill0(g_mi, (long long)N_MI * D2, stream);
    run_layer(Wp1, bp1, Wn1, Ws1, bs1, D1, D2, h_tf, h_tg, h_mi, g_tf, g_tg, g_mi);

    // Mid activation: extra leaky_relu on layer-1 outputs
    {
        long long n;
        n = (long long)N_TF * D2; leaky_ip<<<(unsigned)((n + 255) / 256), 256, 0, stream>>>(g_tf, n);
        n = (long long)N_TG * D2; leaky_ip<<<(unsigned)((n + 255) / 256), 256, 0, stream>>>(g_tg, n);
        n = (long long)N_MI * D2; leaky_ip<<<(unsigned)((n + 255) / 256), 256, 0, stream>>>(g_mi, n);
    }

    // Layer 2 -> d_out (concat order: tf rows, tg rows, mi rows)
    float* outf = (float*)d_out;
    float* o_tf = outf;
    float* o_tg = outf + (size_t)N_TF * D3;
    float* o_mi = outf + (size_t)(N_TF + N_TG) * D3;
    launch_fill0(outf, (long long)(N_TF + N_TG + N_MI) * D3, stream);
    run_layer(Wp2, bp2, Wn2, Ws2, bs2, D2, D3, g_tf, g_tg, g_mi, o_tf, o_tg, o_mi);
}
